// EasyControlNetwork_12343736009327
// MI455X (gfx1250) — matrix operand order repaired and checked by reading, not compiled, not run on hardware
//
#include <hip/hip_runtime.h>
#include <hip/hip_bf16.h>

// Problem constants (from reference): B=2, S_T=2048, S_C=512, H=16, D=128
#define BB   2
#define ST   2048
#define SC   512
#define HH   16
#define DD   128
#define MODEL (HH * DD)        // 2048

typedef __bf16 v16bf __attribute__((ext_vector_type(16)));
typedef __bf16 v4bf  __attribute__((ext_vector_type(4)));
typedef float  v8f   __attribute__((ext_vector_type(8)));
typedef int    v4i   __attribute__((ext_vector_type(4)));

#ifndef __has_builtin
#define __has_builtin(x) 0
#endif

// Async global->LDS bulk copy path (gfx1250 GLOBAL_LOAD_ASYNC_TO_LDS_B128,
// tracked with ASYNCcnt). Probe round 3: builtin exists; param0 is
// 'int4 addrspace(1)*' (global), so use typed int4 pointers.
#if defined(__AMDGCN__) && __has_builtin(__builtin_amdgcn_global_load_async_to_lds_b128)
#define HAVE_ASYNC_LDS 1
typedef __attribute__((address_space(1))) v4i* gptr4;
typedef __attribute__((address_space(3))) v4i* lptr4;
__device__ __forceinline__ void wait_async0() {
#if __has_builtin(__builtin_amdgcn_s_wait_asynccnt)
  __builtin_amdgcn_s_wait_asynccnt(0);
#else
  asm volatile("s_wait_asynccnt 0x0" ::: "memory");
#endif
}
#else
#define HAVE_ASYNC_LDS 0
#endif

// ---------------------------------------------------------------------------
// Fragment loaders (wave32 WMMA layouts, CDNA5 ISA §7.12.2)
// A (16x32 bf16): lanes 0-15 hold row M=lane, K = {kb..kb+7, kb+16..kb+23};
//                 lanes 16-31 same rows, K = {kb+8..kb+15, kb+24..kb+31}.
// B (32x16 bf16): lane holds column N=lane&15, same K order as A:
//                 lanes 0-15 -> K {kb..kb+7, kb+16..kb+23}, lanes 16-31 -> K {kb+8..kb+15, kb+24..kb+31}.
// ---------------------------------------------------------------------------
__device__ __forceinline__ v16bf load_a_frag(const __bf16* base, int row0,
                                             int ldk, int kb) {
  const int lane = threadIdx.x & 31;
  const int l16 = lane & 15;
  const int hi  = lane >> 4;
  const __bf16* p0 = base + (size_t)(row0 + l16) * ldk + kb + hi * 8;
  v16bf r;
#pragma unroll
  for (int i = 0; i < 8; ++i) { r[i] = p0[i]; r[8 + i] = p0[16 + i]; }
  return r;
}

__device__ __forceinline__ v16bf load_b_frag(const __bf16* base, int col0,
                                             int ldk, int kb) {
  const int lane = threadIdx.x & 31;
  const int l16 = lane & 15;
  const int hi  = lane >> 4;
  const __bf16* p0 = base + (size_t)(col0 + l16) * ldk + kb + hi * 8;
  v16bf r;
#pragma unroll
  for (int i = 0; i < 8; ++i) { r[i] = p0[i]; r[8 + i] = p0[16 + i]; }
  return r;
}

__device__ __forceinline__ v8f vzero8() {
  v8f v;
#pragma unroll
  for (int i = 0; i < 8; ++i) v[i] = 0.0f;
  return v;
}

// ---------------------------------------------------------------------------
// One-time fp32 -> bf16 conversion (4 elements / thread, b128 loads).
// ---------------------------------------------------------------------------
__global__ __launch_bounds__(256)
void cvt_kernel(const float* __restrict__ in, __bf16* __restrict__ out,
                int n4) {
  const int i = blockIdx.x * blockDim.x + threadIdx.x;
  if (i < n4) {
    const float4 f = ((const float4*)in)[i];
    v4bf o;
    o[0] = (__bf16)f.x; o[1] = (__bf16)f.y;
    o[2] = (__bf16)f.z; o[3] = (__bf16)f.w;
    *(v4bf*)(out + (size_t)i * 4) = o;
  }
}

// ---------------------------------------------------------------------------
// Flash attention over joint [K_t ; K_c] with logit bias b on cond positions.
// Writes bf16 x = joint_out.reshape(B, S_T, H*D) into workspace.
// Grid: (ST/128, BB*HH), 256 threads (8 waves); wave w owns Q rows w*16..+15.
// KT = float (direct path) or __bf16 (pre-converted path).
// ---------------------------------------------------------------------------
template <typename KT>
__global__ __launch_bounds__(256)
void attn_kernel(const float* __restrict__ q, const KT* __restrict__ k,
                 const KT* __restrict__ v, const KT* __restrict__ ck,
                 const KT* __restrict__ cv, const float* __restrict__ bcond,
                 __bf16* __restrict__ x) {
  __shared__ __bf16 lq[128 * 128];    // Q tile, row-major [row][d]   (32 KB)
  __shared__ __bf16 lk[64 * 128];     // K tile, row-major [key][d]   (16 KB)
  __shared__ __bf16 lvt[128 * 64];    // V tile transposed [d][key]   (16 KB)
  __shared__ __bf16 lp[8 * 16 * 64];  // per-wave P staging           (16 KB)

  const int tid  = threadIdx.x;
  const int wave = tid >> 5;
  const int lane = tid & 31;
  const int l16  = lane & 15;
  const int hi   = lane >> 4;
  const int q0   = blockIdx.x * 128;
  const int b    = blockIdx.y / HH;
  const int h    = blockIdx.y % HH;
  const float scale  = 0.088388347648318447f;   // 1/sqrt(128)
  const float bias_c = bcond[0];

  // Stage Q tile fp32 -> bf16 (once).
  for (int idx = tid; idx < 128 * 128; idx += 256) {
    const int r = idx >> 7, c = idx & 127;
    lq[idx] = (__bf16)q[(((size_t)b * ST + q0 + r) * HH + h) * DD + c];
  }

  v8f o[8];
  float m_[8], l_[8];
#pragma unroll
  for (int i = 0; i < 8; ++i) { o[i] = vzero8(); m_[i] = -1e30f; l_[i] = 0.0f; }

  __bf16* pw = lp + wave * (16 * 64);

  for (int t = 0; t < 40; ++t) {
    const KT* kg; const KT* vg; int kb0; int sk; float bias;
    if (t < 32) { kg = k;  vg = v;  kb0 = t * 64;        sk = ST; bias = 0.0f; }
    else        { kg = ck; vg = cv; kb0 = (t - 32) * 64; sk = SC; bias = bias_c; }

    __syncthreads();
#if HAVE_ASYNC_LDS
    if constexpr (sizeof(KT) == 2) {
      // K tile: pure bf16 copy -> async DMA straight into LDS (no VGPRs).
      for (int idx = tid; idx < 64 * 16; idx += 256) {
        const int r = idx >> 4, ch = idx & 15;   // 16-byte chunks of a row
        const size_t g = (((size_t)b * sk + kb0 + r) * HH + h) * DD + ch * 8;
        __builtin_amdgcn_global_load_async_to_lds_b128(
            (gptr4)(void*)(kg + g), (lptr4)(void*)(lk + r * 128 + ch * 8),
            0, 0);
      }
      // V tile: transpose into [d][key] (scattered 2B stores, manual).
      for (int idx = tid; idx < 64 * 128; idx += 256) {
        const int r = idx >> 7, c = idx & 127;
        lvt[c * 64 + r] =
            (__bf16)vg[(((size_t)b * sk + kb0 + r) * HH + h) * DD + c];
      }
      wait_async0();
    } else
#endif
    {
      for (int idx = tid; idx < 64 * 128; idx += 256) {
        const int r = idx >> 7, c = idx & 127;
        const size_t g = (((size_t)b * sk + kb0 + r) * HH + h) * DD + c;
        lk[idx] = (__bf16)kg[g];
        lvt[c * 64 + r] = (__bf16)vg[g];   // transpose V into [d][key]
      }
    }
    __syncthreads();

    // Prefetch next tile (one cacheline-ish stride per thread) so the
    // HBM/L2 fetch overlaps this tile's 32 WMMAs (global_prefetch path).
    if (t + 1 < 40) {
      const KT* kgn; const KT* vgn; int kbn; int skn;
      if (t + 1 < 32) { kgn = k;  vgn = v;  kbn = (t + 1) * 64;  skn = ST; }
      else            { kgn = ck; vgn = cv; kbn = (t - 31) * 64; skn = SC; }
      const size_t gp =
          (((size_t)b * skn + kbn + (tid >> 2)) * HH + h) * DD + (tid & 3) * 32;
      __builtin_prefetch(&kgn[gp], 0, 1);
      __builtin_prefetch(&vgn[gp], 0, 1);
    }

    // ---- S = Q K^T over this 64-key tile: 4 N-tiles x 4 K-chunks ----
    v8f s[4];
#pragma unroll
    for (int n = 0; n < 4; ++n) s[n] = vzero8();
#pragma unroll
    for (int c = 0; c < 4; ++c) {
      const v16bf a = load_a_frag(lq, wave * 16, 128, c * 32);
      v16bf bf[4];
#pragma unroll
      for (int n = 0; n < 4; ++n) bf[n] = load_b_frag(lk, n * 16, 128, c * 32);
#pragma unroll
      for (int n = 0; n < 4; ++n)
        s[n] = __builtin_amdgcn_wmma_f32_16x16x32_bf16(
            false, a, false, bf[n], (short)0, s[n], false, false);
    }

    // scale + bias, online softmax (rows live per-VGPR across 16-lane halves)
#pragma unroll
    for (int n = 0; n < 4; ++n)
#pragma unroll
      for (int r = 0; r < 8; ++r) s[n][r] = s[n][r] * scale + bias;

#pragma unroll
    for (int r = 0; r < 8; ++r) {
      float tmax = s[0][r];
#pragma unroll
      for (int n = 1; n < 4; ++n) tmax = fmaxf(tmax, s[n][r]);
      tmax = fmaxf(tmax, __shfl_xor(tmax, 1));
      tmax = fmaxf(tmax, __shfl_xor(tmax, 2));
      tmax = fmaxf(tmax, __shfl_xor(tmax, 4));
      tmax = fmaxf(tmax, __shfl_xor(tmax, 8));
      const float mn = fmaxf(m_[r], tmax);
      const float f  = __expf(m_[r] - mn);
      m_[r] = mn;
      l_[r] *= f;
#pragma unroll
      for (int dt = 0; dt < 8; ++dt) o[dt][r] *= f;
      float rs = 0.0f;
#pragma unroll
      for (int n = 0; n < 4; ++n) {
        const float p = __expf(s[n][r] - mn);
        s[n][r] = p;
        rs += p;
      }
      rs += __shfl_xor(rs, 1);
      rs += __shfl_xor(rs, 2);
      rs += __shfl_xor(rs, 4);
      rs += __shfl_xor(rs, 8);
      l_[r] += rs;
    }

    // ---- Re-layout P (C-layout -> A-layout) through per-wave LDS ----
#pragma unroll
    for (int n = 0; n < 4; ++n)
#pragma unroll
      for (int r = 0; r < 8; ++r)
        pw[(hi * 8 + r) * 64 + n * 16 + l16] = (__bf16)s[n][r];

    // ---- O += P V : 2 K-chunks (64 keys) x 8 d-tiles ----
#pragma unroll
    for (int c = 0; c < 2; ++c) {
      const v16bf a = load_a_frag(pw, 0, 64, c * 32);
      v16bf bf[8];
#pragma unroll
      for (int dt = 0; dt < 8; ++dt)
        bf[dt] = load_b_frag(lvt, dt * 16, 64, c * 32);
#pragma unroll
      for (int dt = 0; dt < 8; ++dt)
        o[dt] = __builtin_amdgcn_wmma_f32_16x16x32_bf16(
            false, a, false, bf[dt], (short)0, o[dt], false, false);
    }
  }

  // Normalize and write x[b, s, h*128 + d] as bf16 for the projection GEMM.
#pragma unroll
  for (int r = 0; r < 8; ++r) {
    const float rec = 1.0f / l_[r];
    const int grow = q0 + wave * 16 + hi * 8 + r;
    const size_t rowbase = ((size_t)b * ST + grow) * MODEL + h * DD;
#pragma unroll
    for (int dt = 0; dt < 8; ++dt)
      x[rowbase + dt * 16 + l16] = (__bf16)(o[dt][r] * rec);
  }
}

// ---------------------------------------------------------------------------
// out = x @ w^T : M=4096, N=2048, K=2048. 128x128 C tile per block (8 waves);
// x is bf16 K-contiguous (direct A fragments from global); w fp32 -> bf16
// staged per 64-wide K chunk in LDS (16 KB). B fragments are pre-loaded into
// distinct registers to break the ds_load -> wait(0) -> wmma serial chain.
// ---------------------------------------------------------------------------
__global__ __launch_bounds__(256)
void proj_kernel(const __bf16* __restrict__ x, const float* __restrict__ w,
                 float* __restrict__ out) {
  __shared__ __bf16 lb[128 * 64];
  const int tid  = threadIdx.x;
  const int wave = tid >> 5;
  const int lane = tid & 31;
  const int l16  = lane & 15;
  const int hi   = lane >> 4;
  const int m0   = blockIdx.x * 128;
  const int n0   = blockIdx.y * 128;

  v8f o[8];
#pragma unroll
  for (int i = 0; i < 8; ++i) o[i] = vzero8();

  for (int kc = 0; kc < 32; ++kc) {
    __syncthreads();
    for (int idx = tid; idx < 128 * 64; idx += 256) {
      const int n = idx >> 6, kk = idx & 63;
      lb[idx] = (__bf16)w[(size_t)(n0 + n) * MODEL + kc * 64 + kk];
    }
    __syncthreads();
#pragma unroll
    for (int c = 0; c < 2; ++c) {
      // A fragment straight from global bf16 x (rows are K-contiguous).
      const __bf16* p0 =
          x + (size_t)(m0 + wave * 16 + l16) * MODEL + kc * 64 + c * 32 + hi * 8;
      v16bf a;
#pragma unroll
      for (int i = 0; i < 8; ++i) { a[i] = p0[i]; a[8 + i] = p0[16 + i]; }
      v16bf bf[8];
#pragma unroll
      for (int nt = 0; nt < 8; ++nt)
        bf[nt] = load_b_frag(lb, nt * 16, 64, c * 32);
#pragma unroll
      for (int nt = 0; nt < 8; ++nt)
        o[nt] = __builtin_amdgcn_wmma_f32_16x16x32_bf16(
            false, a, false, bf[nt], (short)0, o[nt], false, false);
    }
  }

#pragma unroll
  for (int nt = 0; nt < 8; ++nt)
#pragma unroll
    for (int r = 0; r < 8; ++r)
      out[(size_t)(m0 + wave * 16 + hi * 8 + r) * MODEL + n0 + nt * 16 + l16] =
          o[nt][r];
}

// ---------------------------------------------------------------------------
extern "C" void kernel_launch(void* const* d_in, const int* in_sizes, int n_in,
                              void* d_out, int out_size, void* d_ws,
                              size_t ws_size, hipStream_t stream) {
  const float* q  = (const float*)d_in[0];
  const float* k  = (const float*)d_in[1];
  const float* v  = (const float*)d_in[2];
  const float* ck = (const float*)d_in[3];
  const float* cv = (const float*)d_in[4];
  const float* bc = (const float*)d_in[5];
  const float* w  = (const float*)d_in[6];
  float* out = (float*)d_out;

  const size_t xElems  = (size_t)BB * ST * MODEL;       // 8.4M  (16 MB bf16)
  const size_t kvElems = (size_t)BB * ST * HH * DD;     // == xElems
  const size_t cElems  = (size_t)BB * SC * HH * DD;     // 2.1M  (4 MB bf16)
  const size_t need = (xElems + 2 * kvElems + 2 * cElems) * sizeof(__bf16);

  __bf16* x = (__bf16*)d_ws;
  const dim3 ga(ST / 128, BB * HH);   // 16 x 32 blocks, 256 threads
  const dim3 gp((BB * ST) / 128, MODEL / 128);  // 32 x 16 blocks

  if (ws_size >= need) {
    // Pre-convert K/V/condK/condV to bf16: halves streamed bytes and turns
    // per-tile LDS staging into a pure 2-byte copy (async DMA eligible).
    __bf16* kb  = x  + xElems;
    __bf16* vb  = kb + kvElems;
    __bf16* ckb = vb + kvElems;
    __bf16* cvb = ckb + cElems;
    const int kv4 = (int)(kvElems / 4), c4 = (int)(cElems / 4);
    cvt_kernel<<<(kv4 + 255) / 256, 256, 0, stream>>>(k,  kb,  kv4);
    cvt_kernel<<<(kv4 + 255) / 256, 256, 0, stream>>>(v,  vb,  kv4);
    cvt_kernel<<<(c4  + 255) / 256, 256, 0, stream>>>(ck, ckb, c4);
    cvt_kernel<<<(c4  + 255) / 256, 256, 0, stream>>>(cv, cvb, c4);
    attn_kernel<__bf16><<<ga, 256, 0, stream>>>(q, kb, vb, ckb, cvb, bc, x);
  } else {
    attn_kernel<float><<<ga, 256, 0, stream>>>(q, k, v, ck, cv, bc, x);
  }

  proj_kernel<<<gp, 256, 0, stream>>>(x, w, out);
}
